// Encoder_Block_77738908057671
// MI455X (gfx1250) — matrix operand order repaired and checked by reading, not compiled, not run on hardware
//
#include <hip/hip_runtime.h>
#include <hip/hip_bf16.h>

typedef __attribute__((ext_vector_type(16))) __bf16 v16bf;
typedef __attribute__((ext_vector_type(8)))  __bf16 v8bf;
typedef __attribute__((ext_vector_type(8)))  float  v8f;

#define SCALE 3.3635856610148585f   // 128^0.25
#define EPS   1e-5f

static __device__ __forceinline__ float wave_sum(float v) {
#pragma unroll
  for (int o = 16; o > 0; o >>= 1) v += __shfl_xor(v, o, 32);
  return v;
}
static __device__ __forceinline__ float wave_max(float v) {
#pragma unroll
  for (int o = 16; o > 0; o >>= 1) v = fmaxf(v, __shfl_xor(v, o, 32));
  return v;
}

// ---- fragment loaders -------------------------------------------------------
// A (16x32, M x K) ISA layout: lanes 0-15 M=lane, K 0..7 then 16..23; lanes 16-31 K 8..15, 24..31.
// fp32 row-major source:
static __device__ __forceinline__ v16bf load_a_frag_f32(const float* __restrict__ A, int lda, int k0) {
  int lane = threadIdx.x & 31;
  int m  = lane & 15;
  int kb = (lane >> 4) * 8;
  v16bf a;
#pragma unroll
  for (int e = 0; e < 16; ++e) {
    int r = e >> 1, h = e & 1;
    int kk = (r < 4) ? (kb + 2 * r + h) : (16 + kb + 2 * (r - 4) + h);
    a[e] = (__bf16)A[(size_t)m * lda + k0 + kk];
  }
  return a;
}
// bf16 row-major source: two contiguous 16B groups per lane.
static __device__ __forceinline__ v16bf load_a_frag_bf(const __bf16* __restrict__ A, int lda, int k0) {
  int lane = threadIdx.x & 31;
  int m  = lane & 15;
  int kb = (lane >> 4) * 8;
  const __bf16* p = A + (size_t)m * lda + k0 + kb;
  v8bf g0 = *(const v8bf*)(p);
  v8bf g1 = *(const v8bf*)(p + 16);
  return __builtin_shufflevector(g0, g1, 0,1,2,3,4,5,6,7,8,9,10,11,12,13,14,15);
}
// B (32x16, K x N) from bf16 *transposed* weights (Nout-major, stride K):
// lane: N = lane%16, K = (lane/16)*8 + e (e<8), 16 + (lane/16)*8 + (e-8) (e>=8) -> two 16B loads.
static __device__ __forceinline__ v16bf load_b_frag_t(const __bf16* __restrict__ WT, int K, int k0, int n0) {
  int lane = threadIdx.x & 31;
  int n = lane & 15;
  int kbase = (lane >> 4) * 8;
  const __bf16* p = WT + (size_t)(n0 + n) * K + k0 + kbase;
  v8bf g0 = *(const v8bf*)(p);
  v8bf g1 = *(const v8bf*)(p + 16);
  return __builtin_shufflevector(g0, g1, 0,1,2,3,4,5,6,7,8,9,10,11,12,13,14,15);
}

// ---- weight convert + transpose: W (KxN fp32) -> WT (NxK bf16) --------------
__global__ void cvt_wt_kernel(const float* __restrict__ W, __bf16* __restrict__ WT, int K, int N) {
  int idx = blockIdx.x * 256 + threadIdx.x;
  if (idx >= K * N) return;
  int n = idx / K, k = idx - n * K;
  WT[(size_t)n * K + k] = (__bf16)W[(size_t)k * N + n];
}

// ---- LayerNorm over rows of 128; fp32 + bf16 outputs ------------------------
__global__ void ln_rows_kernel(const float* __restrict__ x, const float* __restrict__ g,
                               const float* __restrict__ b, float* __restrict__ outf,
                               __bf16* __restrict__ outh, int rows) {
  int wave = threadIdx.x >> 5, lane = threadIdx.x & 31;
  int row = blockIdx.x * 8 + wave;
  if (row >= rows) return;
  const float* xr = x + (size_t)row * 128;
  float v[4], s = 0.f, s2 = 0.f;
#pragma unroll
  for (int j = 0; j < 4; ++j) { v[j] = xr[lane + 32 * j]; s += v[j]; s2 += v[j] * v[j]; }
  s = wave_sum(s); s2 = wave_sum(s2);
  float mu = s * (1.f / 128.f);
  float var = s2 * (1.f / 128.f) - mu * mu;
  float rs = rsqrtf(var + EPS);
#pragma unroll
  for (int j = 0; j < 4; ++j) {
    int c = lane + 32 * j;
    float o = (v[j] - mu) * rs * g[c] + b[c];
    outf[(size_t)row * 128 + c] = o;
    outh[(size_t)row * 128 + c] = (__bf16)o;
  }
}

// ---- GEMM (bf16 A, bf16T W) + bias -> fp32 ----------------------------------
__global__ void gemm_bias_kernel(const __bf16* __restrict__ A, const __bf16* __restrict__ WT,
                                 const float* __restrict__ bias, float* __restrict__ out, int M) {
  int wave = threadIdx.x >> 5, lane = threadIdx.x & 31;
  int row0 = blockIdx.x * 16;
  int col0 = wave * 16;
  const __bf16* Ar = A + (size_t)row0 * 128;
  v8f acc = {};
#pragma unroll
  for (int kc = 0; kc < 4; ++kc) {
    v16bf a = load_a_frag_bf(Ar, 128, kc * 32);
    v16bf b = load_b_frag_t(WT, 128, kc * 32, col0);
    acc = __builtin_amdgcn_wmma_f32_16x16x32_bf16(false, a, false, b, (short)0, acc, false, false);
  }
  int n = lane & 15, mb = (lane >> 4) * 8;
#pragma unroll
  for (int j = 0; j < 8; ++j)
    out[(size_t)(row0 + mb + j) * 128 + col0 + n] = acc[j] + bias[col0 + n];
}

// ---- scores + dual softmax + message (fp32 out + bf16 copy), per (b,h) ------
__global__ void scores_kernel(const float* __restrict__ q, const float* __restrict__ k,
                              float* __restrict__ msg, __bf16* __restrict__ msgh) {
  extern __shared__ float sm[];
  float* S    = sm;            // 128*128
  float* qs   = S + 16384;     // 128*16
  float* rmax = qs + 2048;
  float* rsum = rmax + 128;
  float* cmax = rsum + 128;
  float* csum = cmax + 128;
  int bh = blockIdx.x, tid = threadIdx.x;
  size_t qbase = (size_t)bh * 2048;
  size_t kbase = (size_t)bh * 128 * 2048;
  for (int i = tid; i < 2048; i += 256) qs[i] = q[qbase + i];
  __syncthreads();
  for (int idx = tid; idx < 16384; idx += 256) {
    int m = idx >> 7, n = idx & 127;
    const float* kr = k + kbase + (size_t)m * 2048 + n * 16;
    const float* qr = qs + m * 16;
    float d = 0.f;
#pragma unroll
    for (int dd = 0; dd < 16; ++dd) d += qr[dd] * kr[dd];
    S[idx] = d * SCALE;
  }
  __syncthreads();
  int wave = tid >> 5, lane = tid & 31;
  for (int r = wave; r < 128; r += 8) {          // row softmax stats (out_attn)
    float mx = -3.4e38f;
#pragma unroll
    for (int j = 0; j < 4; ++j) mx = fmaxf(mx, S[r * 128 + lane + 32 * j]);
    mx = wave_max(mx);
    float sme = 0.f;
#pragma unroll
    for (int j = 0; j < 4; ++j) sme += __expf(S[r * 128 + lane + 32 * j] - mx);
    sme = wave_sum(sme);
    if (lane == 0) { rmax[r] = mx; rsum[r] = sme; }
  }
  for (int c = wave; c < 128; c += 8) {          // column softmax stats (in_attn)
    float mx = -3.4e38f;
#pragma unroll
    for (int j = 0; j < 4; ++j) mx = fmaxf(mx, S[(lane + 32 * j) * 128 + c]);
    mx = wave_max(mx);
    float sme = 0.f;
#pragma unroll
    for (int j = 0; j < 4; ++j) sme += __expf(S[(lane + 32 * j) * 128 + c] - mx);
    sme = wave_sum(sme);
    if (lane == 0) { cmax[c] = mx; csum[c] = sme; }
  }
  __syncthreads();
  float* mo = msg + (size_t)bh * 16384;
  __bf16* mh = msgh + (size_t)bh * 16384;
  for (int idx = tid; idx < 16384; idx += 256) {
    int i = idx >> 7, j = idx & 127;
    float outp = (i == j) ? 0.f : __expf(S[i * 128 + j] - rmax[i]) / rsum[i];
    float inp  = __expf(S[j * 128 + i] - cmax[i]) / csum[i];
    float m = outp + inp;
    mo[idx] = m;
    mh[idx] = (__bf16)m;
  }
}

// ---- message(128x128 bf16) @ rhs(128x16 fp32, LDS-staged); A=1 node, 128 edge
__global__ void msg_matmul_kernel(const __bf16* __restrict__ msgh, const float* __restrict__ rhs,
                                  float* __restrict__ out, int A) {
  __shared__ float rs_s[2048];
  int blk = blockIdx.x, tid = threadIdx.x;
  int wave = tid >> 5, lane = tid & 31;
  size_t mbase = (size_t)(blk / A) * 16384;
  size_t rbase = (size_t)blk * 2048;
  for (int i = tid; i < 2048; i += 256) rs_s[i] = rhs[rbase + i];
  __syncthreads();
  int m0 = wave * 16;
  int n = lane & 15, kbase = (lane >> 4) * 8;
  v8f acc = {};
#pragma unroll
  for (int kc = 0; kc < 4; ++kc) {
    v16bf a = load_a_frag_bf(msgh + mbase + (size_t)m0 * 128, 128, kc * 32);
    v16bf b;
#pragma unroll
    for (int e = 0; e < 16; ++e)
      b[e] = (__bf16)rs_s[(kc * 32 + kbase + e + (e >> 3) * 8) * 16 + n];
    acc = __builtin_amdgcn_wmma_f32_16x16x32_bf16(false, a, false, b, (short)0, acc, false, false);
  }
  int mb = (lane >> 4) * 8;
#pragma unroll
  for (int j = 0; j < 8; ++j)
    out[rbase + (size_t)(m0 + mb + j) * 16 + n] = acc[j];
}

// ---- GEMM (fp32 A, bf16T W) + bias + residual + LayerNorm; fp32 + bf16 out --
__global__ void gemm_ln_kernel(const float* __restrict__ A, const __bf16* __restrict__ WT,
                               const float* __restrict__ bias, const float* __restrict__ res,
                               const float* __restrict__ g, const float* __restrict__ b,
                               float* __restrict__ outf, __bf16* __restrict__ outh, int M) {
  __shared__ float tile[16 * 128];
  int wave = threadIdx.x >> 5, lane = threadIdx.x & 31;
  int row0 = blockIdx.x * 16;
  int col0 = wave * 16;
  const float* Ar = A + (size_t)row0 * 128;
  v8f acc = {};
#pragma unroll
  for (int kc = 0; kc < 4; ++kc) {
    v16bf a = load_a_frag_f32(Ar, 128, kc * 32);
    v16bf bb = load_b_frag_t(WT, 128, kc * 32, col0);
    acc = __builtin_amdgcn_wmma_f32_16x16x32_bf16(false, a, false, bb, (short)0, acc, false, false);
  }
  int n = lane & 15, mb = (lane >> 4) * 8;
#pragma unroll
  for (int j = 0; j < 8; ++j)
    tile[(mb + j) * 128 + col0 + n] = acc[j] + bias[col0 + n];
  __syncthreads();
#pragma unroll
  for (int rr = 0; rr < 2; ++rr) {
    int r = wave * 2 + rr;
    float v[4], s = 0.f, s2 = 0.f;
#pragma unroll
    for (int j = 0; j < 4; ++j) {
      int c = lane + 32 * j;
      v[j] = tile[r * 128 + c] + res[(size_t)(row0 + r) * 128 + c];
      s += v[j]; s2 += v[j] * v[j];
    }
    s = wave_sum(s); s2 = wave_sum(s2);
    float mu = s * (1.f / 128.f), var = s2 * (1.f / 128.f) - mu * mu, rs = rsqrtf(var + EPS);
#pragma unroll
    for (int j = 0; j < 4; ++j) {
      int c = lane + 32 * j;
      float o = (v[j] - mu) * rs * g[c] + b[c];
      outf[(size_t)(row0 + r) * 128 + c] = o;
      outh[(size_t)(row0 + r) * 128 + c] = (__bf16)o;
    }
  }
}

// ---- fused MLP: sigmoid(Xh@W1T+b1)@W2T+b2, residual Xf, LayerNorm -----------
__global__ void mlp_ln_kernel(const __bf16* __restrict__ Xh, const float* __restrict__ Xf,
                              const __bf16* __restrict__ W1T, const float* __restrict__ b1,
                              const __bf16* __restrict__ W2T, const float* __restrict__ b2,
                              const float* __restrict__ g, const float* __restrict__ bt,
                              float* __restrict__ out, int M) {
  __shared__ __bf16 Hs[16 * 512];
  __shared__ float tile[16 * 128];
  int wave = threadIdx.x >> 5, lane = threadIdx.x & 31;
  int row0 = blockIdx.x * 16;
  const __bf16* Xr = Xh + (size_t)row0 * 128;
  // phase 1: hidden (16x512) bf16 into LDS
#pragma unroll
  for (int t = 0; t < 4; ++t) {
    int col0 = (wave + t * 8) * 16;
    v8f acc = {};
#pragma unroll
    for (int kc = 0; kc < 4; ++kc) {
      v16bf a = load_a_frag_bf(Xr, 128, kc * 32);
      v16bf bb = load_b_frag_t(W1T, 128, kc * 32, col0);
      acc = __builtin_amdgcn_wmma_f32_16x16x32_bf16(false, a, false, bb, (short)0, acc, false, false);
    }
    int n = lane & 15, mb = (lane >> 4) * 8;
#pragma unroll
    for (int j = 0; j < 8; ++j) {
      float hv = acc[j] + b1[col0 + n];
      hv = 1.f / (1.f + __expf(-hv));
      Hs[(mb + j) * 512 + col0 + n] = (__bf16)hv;
    }
  }
  __syncthreads();
  // phase 2: out = hidden @ W2T + b2, K=512; A frags via vector LDS loads
  int col0 = wave * 16;
  int m = lane & 15, kb = (lane >> 4) * 8;
  v8f acc = {};
#pragma unroll
  for (int kc = 0; kc < 16; ++kc) {
    const __bf16* hp = Hs + m * 512 + kc * 32 + kb;
    v8bf g0 = *(const v8bf*)(hp);
    v8bf g1 = *(const v8bf*)(hp + 16);
    v16bf a = __builtin_shufflevector(g0, g1, 0,1,2,3,4,5,6,7,8,9,10,11,12,13,14,15);
    v16bf bb = load_b_frag_t(W2T, 512, kc * 32, col0);
    acc = __builtin_amdgcn_wmma_f32_16x16x32_bf16(false, a, false, bb, (short)0, acc, false, false);
  }
  int n = lane & 15, mb = (lane >> 4) * 8;
#pragma unroll
  for (int j = 0; j < 8; ++j)
    tile[(mb + j) * 128 + col0 + n] = acc[j] + b2[col0 + n];
  __syncthreads();
#pragma unroll
  for (int rr = 0; rr < 2; ++rr) {
    int r = wave * 2 + rr;
    float v[4], s = 0.f, s2 = 0.f;
#pragma unroll
    for (int j = 0; j < 4; ++j) {
      int c = lane + 32 * j;
      v[j] = tile[r * 128 + c] + Xf[(size_t)(row0 + r) * 128 + c];
      s += v[j]; s2 += v[j] * v[j];
    }
    s = wave_sum(s); s2 = wave_sum(s2);
    float mu = s * (1.f / 128.f), var = s2 * (1.f / 128.f) - mu * mu, rs = rsqrtf(var + EPS);
#pragma unroll
    for (int j = 0; j < 4; ++j) {
      int c = lane + 32 * j;
      out[(size_t)(row0 + r) * 128 + c] = (v[j] - mu) * rs * g[c] + bt[c];
    }
  }
}

extern "C" void kernel_launch(void* const* d_in, const int* in_sizes, int n_in,
                              void* d_out, int out_size, void* d_ws, size_t ws_size,
                              hipStream_t stream) {
  (void)in_sizes; (void)n_in; (void)out_size; (void)ws_size;
  const float* x   = (const float*)d_in[0];
  const float* y   = (const float*)d_in[1];
  const float* qw  = (const float*)d_in[2];
  const float* qb  = (const float*)d_in[3];
  const float* kw  = (const float*)d_in[4];
  const float* kb  = (const float*)d_in[5];
  const float* vw  = (const float*)d_in[6];
  const float* vb  = (const float*)d_in[7];
  const float* onw = (const float*)d_in[8];
  const float* onb = (const float*)d_in[9];
  const float* oew = (const float*)d_in[10];
  const float* oeb = (const float*)d_in[11];
  const float* ln1g = (const float*)d_in[12]; const float* ln1b = (const float*)d_in[13];
  const float* ln2g = (const float*)d_in[14]; const float* ln2b = (const float*)d_in[15];
  const float* ln3g = (const float*)d_in[16]; const float* ln3b = (const float*)d_in[17];
  const float* ln4g = (const float*)d_in[18]; const float* ln4b = (const float*)d_in[19];
  const float* ln5g = (const float*)d_in[20]; const float* ln5b = (const float*)d_in[21];
  const float* ln6g = (const float*)d_in[22]; const float* ln6b = (const float*)d_in[23];
  const float* m1w1 = (const float*)d_in[24]; const float* m1b1 = (const float*)d_in[25];
  const float* m1w2 = (const float*)d_in[26]; const float* m1b2 = (const float*)d_in[27];
  const float* m2w1 = (const float*)d_in[28]; const float* m2b1 = (const float*)d_in[29];
  const float* m2w2 = (const float*)d_in[30]; const float* m2b2 = (const float*)d_in[31];

  const size_t NX = (size_t)16 * 128 * 128;          // 262144
  const size_t NY = (size_t)16 * 128 * 128 * 128;    // 33554432
  const size_t NM = (size_t)16 * 8 * 128 * 128;      // 2097152 (message)

  float* wsf = (float*)d_ws;
  float* x1f    = wsf; wsf += NX;
  float* q      = wsf; wsf += NX;
  float* v      = wsf; wsf += NX;
  float* node_h = wsf; wsf += NX;
  float* x2f    = wsf; wsf += NX;
  float* y1f    = wsf; wsf += NY;   // becomes y2 (fp32) in place
  float* kk     = wsf; wsf += NY;
  float* eh     = wsf; wsf += NY;
  __bf16* wsh = (__bf16*)wsf;
  __bf16* x1h  = wsh; wsh += NX;
  __bf16* x2h  = wsh; wsh += NX;
  __bf16* y1h  = wsh; wsh += NY;
  __bf16* y2h  = wsh; wsh += NY;
  __bf16* msgh = wsh; wsh += NM;
  __bf16* qwT  = wsh; wsh += 128 * 128;
  __bf16* kwT  = wsh; wsh += 128 * 128;
  __bf16* vwT  = wsh; wsh += 128 * 128;
  __bf16* onwT = wsh; wsh += 128 * 128;
  __bf16* oewT = wsh; wsh += 128 * 128;
  __bf16* m1w1T = wsh; wsh += 128 * 512;
  __bf16* m1w2T = wsh; wsh += 512 * 128;
  __bf16* m2w1T = wsh; wsh += 128 * 512;
  __bf16* m2w2T = wsh; wsh += 512 * 128;

  float* xo  = (float*)d_out;
  float* yo  = xo + NX;
  float* msg = yo + NY;   // third output: message (B,H,N,N)

  // weight convert+transpose (tiny, one-time per call)
  cvt_wt_kernel<<<64, 256, 0, stream>>>(qw, qwT, 128, 128);
  cvt_wt_kernel<<<64, 256, 0, stream>>>(kw, kwT, 128, 128);
  cvt_wt_kernel<<<64, 256, 0, stream>>>(vw, vwT, 128, 128);
  cvt_wt_kernel<<<64, 256, 0, stream>>>(onw, onwT, 128, 128);
  cvt_wt_kernel<<<64, 256, 0, stream>>>(oew, oewT, 128, 128);
  cvt_wt_kernel<<<256, 256, 0, stream>>>(m1w1, m1w1T, 128, 512);
  cvt_wt_kernel<<<256, 256, 0, stream>>>(m1w2, m1w2T, 512, 128);
  cvt_wt_kernel<<<256, 256, 0, stream>>>(m2w1, m2w1T, 128, 512);
  cvt_wt_kernel<<<256, 256, 0, stream>>>(m2w2, m2w2T, 512, 128);

  const int MX = 2048, MY = 262144;
  ln_rows_kernel<<<MX / 8, 256, 0, stream>>>(x, ln1g, ln1b, x1f, x1h, MX);
  ln_rows_kernel<<<MY / 8, 256, 0, stream>>>(y, ln2g, ln2b, y1f, y1h, MY);
  gemm_bias_kernel<<<MX / 16, 256, 0, stream>>>(x1h, qwT, qb, q, MX);
  gemm_bias_kernel<<<MX / 16, 256, 0, stream>>>(x1h, vwT, vb, v, MX);
  gemm_bias_kernel<<<MY / 16, 256, 0, stream>>>(y1h, kwT, kb, kk, MY);
  size_t smem = (16384 + 2048 + 4 * 128) * sizeof(float);
  scores_kernel<<<128, 256, smem, stream>>>(q, kk, msg, msgh);
  msg_matmul_kernel<<<128, 256, 0, stream>>>(msgh, v, node_h, 1);
  msg_matmul_kernel<<<16384, 256, 0, stream>>>(msgh, kk, eh, 128);
  gemm_ln_kernel<<<MX / 16, 256, 0, stream>>>(node_h, onwT, onb, x1f, ln3g, ln3b, x2f, x2h, MX);
  gemm_ln_kernel<<<MY / 16, 256, 0, stream>>>(eh, oewT, oeb, y1f, ln4g, ln4b, y1f, y2h, MY);
  mlp_ln_kernel<<<MX / 16, 256, 0, stream>>>(x2h, x2f, m1w1T, m1b1, m1w2T, m1b2, ln5g, ln5b, xo, MX);
  mlp_ln_kernel<<<MY / 16, 256, 0, stream>>>(y2h, y1f, m2w1T, m2b1, m2w2T, m2b2, ln6g, ln6b, yo, MY);
}
